// SignLanguageLSTM_85624468013559
// MI455X (gfx1250) — matrix operand order repaired and checked by reading, not compiled, not run on hardware
//
#include <hip/hip_runtime.h>
#include <hip/hip_bf16.h>
#include <math.h>

// ---------------------------------------------------------------------------
// SignLanguageLSTM for MI455X (gfx1250): bf16 WMMA GEMMs + per-step WMMA
// recurrence, fused attention epilogue. wave32 / v_wmma_f32_16x16x32_bf16.
// ---------------------------------------------------------------------------

typedef __bf16 v16bf __attribute__((ext_vector_type(16)));
typedef __bf16 v8bf  __attribute__((ext_vector_type(8)));
typedef float  v8f   __attribute__((ext_vector_type(8)));

#define BDIM  128
#define TDIM  256
#define INDIM 1662
#define INPAD 1664   // padded to multiple of 32 for WMMA K-steps
#define HDIM  512
#define H2    1024
#define G4    2048
#define NCLS  100
#define BT    (BDIM * TDIM)

// ---- WMMA fragment loaders (ISA 7.12.2 layouts, K-contiguous memory) ------
// A: 16-bit 16x32 (MxK). lane m = lane&15; halves hold K in two 8-wide runs:
//   k = (i>>3)*16 + (lane>>4)*8 + (i&7)  -> two contiguous 16B loads.
__device__ __forceinline__ v16bf load_a_frag(const __bf16* __restrict__ A,
                                             int lda, int m0, int k0, int lane) {
  const __bf16* row = A + (size_t)(m0 + (lane & 15)) * lda + (k0 + ((lane >> 4) << 3));
  v8bf lo = *reinterpret_cast<const v8bf*>(row);
  v8bf hi = *reinterpret_cast<const v8bf*>(row + 16);
  return __builtin_shufflevector(lo, hi, 0,1,2,3,4,5,6,7,8,9,10,11,12,13,14,15);
}
// B: 16-bit 32x16 (KxN). lane n = lane&15; halves i=0..15 hold
//   k = (i>>3)*16 + (lane>>4)*8 + (i&7)  -> two contiguous 16B loads from row-major [N,K].
__device__ __forceinline__ v16bf load_b_frag(const __bf16* __restrict__ W,
                                             int ldw, int n0, int k0, int lane) {
  const __bf16* p = W + (size_t)(n0 + (lane & 15)) * ldw + (k0 + ((lane >> 4) << 3));
  v8bf lo = *reinterpret_cast<const v8bf*>(p);
  v8bf hi = *reinterpret_cast<const v8bf*>(p + 16);
  return __builtin_shufflevector(lo, hi, 0,1,2,3,4,5,6,7,8,9,10,11,12,13,14,15);
}
__device__ __forceinline__ v8f wmma_bf16(v16bf a, v16bf b, v8f c) {
  return __builtin_amdgcn_wmma_f32_16x16x32_bf16(false, a, false, b, (short)0, c,
                                                 false, false);
}
__device__ __forceinline__ float sigm(float x) { return 1.0f / (1.0f + expf(-x)); }

// ---------------------------------------------------------------------------
// Generic WMMA GEMM:  C[M,N] = A[M,K](bf16) * W[N,K]^T(bf16) + bias[N]
// Block tile 128x128, 8 waves in 4(M)x2(N); wave = 2 M-tiles x 4 N-tiles.
// MODE 0: store C + bias.
// MODE 1: fused attention: s = tanh(acc + ab1[n]) * aw2[n], row-reduced
//         into scores[m] via atomics (tanh tensor never stored).
// ---------------------------------------------------------------------------
template <int MODE>
__global__ __launch_bounds__(256) void wmma_gemm_k(
    const __bf16* __restrict__ A, int lda,
    const __bf16* __restrict__ W, int ldw,
    const float* __restrict__ bias,
    float* __restrict__ C, int N, int K,
    const float* __restrict__ aw2,
    float* __restrict__ scores) {
  const int lane = threadIdx.x & 31;
  const int wave = threadIdx.x >> 5;
  const int m0 = blockIdx.x * 128 + (wave >> 1) * 32;
  const int n0 = blockIdx.y * 128 + (wave & 1) * 64;

  v8f acc[2][4] = {};
  for (int k = 0; k < K; k += 32) {
    if (k + 32 < K) {  // gfx1250 global_prefetch_b8
      __builtin_prefetch(A + (size_t)(m0 + (lane & 15)) * lda + k + 32, 0, 0);
      __builtin_prefetch(W + (size_t)(n0 + (lane & 15)) * ldw + k + 32, 0, 0);
    }
    v16bf a0 = load_a_frag(A, lda, m0, k, lane);
    v16bf a1 = load_a_frag(A, lda, m0 + 16, k, lane);
    v16bf b0 = load_b_frag(W, ldw, n0, k, lane);
    v16bf b1 = load_b_frag(W, ldw, n0 + 16, k, lane);
    v16bf b2 = load_b_frag(W, ldw, n0 + 32, k, lane);
    v16bf b3 = load_b_frag(W, ldw, n0 + 48, k, lane);
    acc[0][0] = wmma_bf16(a0, b0, acc[0][0]);
    acc[0][1] = wmma_bf16(a0, b1, acc[0][1]);
    acc[0][2] = wmma_bf16(a0, b2, acc[0][2]);
    acc[0][3] = wmma_bf16(a0, b3, acc[0][3]);
    acc[1][0] = wmma_bf16(a1, b0, acc[1][0]);
    acc[1][1] = wmma_bf16(a1, b1, acc[1][1]);
    acc[1][2] = wmma_bf16(a1, b2, acc[1][2]);
    acc[1][3] = wmma_bf16(a1, b3, acc[1][3]);
  }

  // C/D layout: VGPR r -> m_local = (lane>>4)*8 + r, n_local = lane&15
  if (MODE == 0) {
#pragma unroll
    for (int i = 0; i < 2; i++)
#pragma unroll
      for (int j = 0; j < 4; j++)
#pragma unroll
        for (int r = 0; r < 8; r++) {
          int m = m0 + i * 16 + ((lane >> 4) << 3) + r;
          int n = n0 + j * 16 + (lane & 15);
          C[(size_t)m * N + n] = acc[i][j][r] + bias[n];
        }
  } else {
#pragma unroll
    for (int i = 0; i < 2; i++)
#pragma unroll
      for (int r = 0; r < 8; r++) {
        int m = m0 + i * 16 + ((lane >> 4) << 3) + r;
        float s = 0.f;
#pragma unroll
        for (int j = 0; j < 4; j++) {
          int n = n0 + j * 16 + (lane & 15);
          s += tanhf(acc[i][j][r] + bias[n]) * aw2[n];
        }
        atomicAdd(&scores[m], s);
      }
  }
}

// ---------------------------------------------------------------------------
// One LSTM timestep, both directions in one launch (grid = 16 x 2).
// Block: hidden slice of 32 -> gate cols {g*512+hd0 .. +31} for g=i,f,g,o.
// Wave w: gate g=w>>1, batch half m0=(w&1)*64 -> 4 M-tiles x 2 N-tiles WMMA,
// K=512 over previous h (double-buffered hin/hout to avoid cross-block RAW).
// Gates staged in 64KB LDS, then fused sigmoid/tanh cell update.
// ---------------------------------------------------------------------------
__global__ __launch_bounds__(256) void lstm_step_k(
    const float* __restrict__ xpF, const float* __restrict__ xpB,
    const __bf16* __restrict__ whhF, const __bf16* __restrict__ whhB,
    const __bf16* __restrict__ hin, __bf16* __restrict__ hout,
    float* __restrict__ cstate,
    __bf16* __restrict__ seq_bf, float* __restrict__ seq_f32, int t) {
  const int dir = blockIdx.y;
  const float* xp = dir ? xpB : xpF;
  const __bf16* whh = dir ? whhB : whhF;
  const __bf16* h = hin + (size_t)dir * (BDIM * HDIM);
  float* c = cstate + (size_t)dir * (BDIM * HDIM);
  const int time = dir ? (TDIM - 1 - t) : t;
  const int hd0 = blockIdx.x * 32;

  __shared__ float z[4][BDIM][32];  // 64 KB of the 320 KB WGP LDS

  const int lane = threadIdx.x & 31;
  const int wave = threadIdx.x >> 5;
  const int g = wave >> 1;
  const int m0 = (wave & 1) * 64;
  const int n0 = g * HDIM + hd0;

  v8f acc[4][2] = {};
  for (int k = 0; k < HDIM; k += 32) {
    v16bf b0 = load_b_frag(whh, HDIM, n0, k, lane);
    v16bf b1 = load_b_frag(whh, HDIM, n0 + 16, k, lane);
#pragma unroll
    for (int i = 0; i < 4; i++) {
      v16bf a = load_a_frag(h, HDIM, m0 + i * 16, k, lane);
      acc[i][0] = wmma_bf16(a, b0, acc[i][0]);
      acc[i][1] = wmma_bf16(a, b1, acc[i][1]);
    }
  }

#pragma unroll
  for (int i = 0; i < 4; i++)
#pragma unroll
    for (int j = 0; j < 2; j++)
#pragma unroll
      for (int r = 0; r < 8; r++) {
        int m = m0 + i * 16 + ((lane >> 4) << 3) + r;
        int nl = j * 16 + (lane & 15);
        z[g][m][nl] = acc[i][j][r] + xp[((size_t)m * TDIM + time) * G4 + n0 + nl];
      }
  __syncthreads();

  for (int e = threadIdx.x; e < BDIM * 32; e += 256) {
    int m = e >> 5, jj = e & 31;
    int hidx = hd0 + jj;
    float iv = z[0][m][jj], fv = z[1][m][jj], gv = z[2][m][jj], ov = z[3][m][jj];
    int ci = m * HDIM + hidx;
    float cn = sigm(fv) * c[ci] + sigm(iv) * tanhf(gv);
    float hn = sigm(ov) * tanhf(cn);
    c[ci] = cn;
    hout[(size_t)dir * (BDIM * HDIM) + ci] = (__bf16)hn;
    size_t so = ((size_t)m * TDIM + time) * H2 + dir * HDIM + hidx;
    seq_bf[so] = (__bf16)hn;
    if (seq_f32) seq_f32[so] = hn;
  }
}

// ---------------------------- small helper kernels -------------------------
__global__ void cvt_pad_k(const float* __restrict__ src, __bf16* __restrict__ dst,
                          int rows, int scols, int dcols) {
  size_t idx = (size_t)blockIdx.x * 256 + threadIdx.x;
  if (idx < (size_t)rows * dcols) {
    int r = (int)(idx / dcols), cc = (int)(idx % dcols);
    float v = (cc < scols) ? src[(size_t)r * scols + cc] : 0.f;
    dst[idx] = (__bf16)v;
  }
}
__global__ void cvt_T_k(const float* __restrict__ src, __bf16* __restrict__ dst,
                        int K, int N) {  // src [K,N] -> dst [N,K] bf16
  size_t idx = (size_t)blockIdx.x * 256 + threadIdx.x;
  if (idx < (size_t)K * N) {
    int n = (int)(idx / K), k = (int)(idx % K);
    dst[(size_t)n * K + k] = (__bf16)src[(size_t)k * N + n];
  }
}
__global__ void zero_f32_k(float* p, int n) {
  int i = blockIdx.x * 256 + threadIdx.x;
  if (i < n) p[i] = 0.f;
}
__global__ void zero_bf16_k(__bf16* p, int n) {
  int i = blockIdx.x * 256 + threadIdx.x;
  if (i < n) p[i] = (__bf16)0.f;
}
__global__ __launch_bounds__(256) void softmax_k(const float* __restrict__ scores,
                                                 float* __restrict__ attw_ws,
                                                 float* __restrict__ attw_out) {
  int b = blockIdx.x, t = threadIdx.x;
  __shared__ float red[TDIM];
  float s = scores[b * TDIM + t];
  red[t] = s;
  __syncthreads();
  for (int o = TDIM / 2; o > 0; o >>= 1) {
    if (t < o) red[t] = fmaxf(red[t], red[t + o]);
    __syncthreads();
  }
  float mx = red[0];
  __syncthreads();
  float e = expf(s - mx);
  red[t] = e;
  __syncthreads();
  for (int o = TDIM / 2; o > 0; o >>= 1) {
    if (t < o) red[t] += red[t + o];
    __syncthreads();
  }
  float w = e / red[0];
  attw_ws[b * TDIM + t] = w;
  attw_out[b * TDIM + t] = w;
}
__global__ __launch_bounds__(256) void context_k(const float* __restrict__ attw,
                                                 const float* __restrict__ lstm,
                                                 float* __restrict__ ctx) {
  int b = blockIdx.x;
  for (int d = threadIdx.x; d < H2; d += 256) {
    float s = 0.f;
    for (int t = 0; t < TDIM; t++)
      s += attw[b * TDIM + t] * lstm[((size_t)b * TDIM + t) * H2 + d];
    ctx[(size_t)b * H2 + d] = s;
  }
}
__global__ __launch_bounds__(256) void classifier_k(
    const float* __restrict__ ctx, const float* __restrict__ cw1,
    const float* __restrict__ cb1, const float* __restrict__ cw2,
    const float* __restrict__ cb2, float* __restrict__ logits) {
  int b = blockIdx.x;
  __shared__ float hid[HDIM];
  for (int j = threadIdx.x; j < HDIM; j += 256) {
    float a = cb1[j];
    for (int k = 0; k < H2; k++) a += ctx[(size_t)b * H2 + k] * cw1[(size_t)k * HDIM + j];
    hid[j] = fmaxf(a, 0.f);
  }
  __syncthreads();
  for (int j = threadIdx.x; j < NCLS; j += 256) {
    float a = cb2[j];
    for (int k = 0; k < HDIM; k++) a += hid[k] * cw2[(size_t)k * NCLS + j];
    logits[b * NCLS + j] = a;
  }
}

// ---------------------------------------------------------------------------
extern "C" void kernel_launch(void* const* d_in, const int* in_sizes, int n_in,
                              void* d_out, int out_size, void* d_ws, size_t ws_size,
                              hipStream_t stream) {
  const float* x     = (const float*)d_in[0];
  const float* wih0f = (const float*)d_in[1];
  const float* whh0f = (const float*)d_in[2];
  const float* b0f   = (const float*)d_in[3];
  const float* wih0b = (const float*)d_in[4];
  const float* whh0b = (const float*)d_in[5];
  const float* b0b   = (const float*)d_in[6];
  const float* wih1f = (const float*)d_in[7];
  const float* whh1f = (const float*)d_in[8];
  const float* b1f   = (const float*)d_in[9];
  const float* wih1b = (const float*)d_in[10];
  const float* whh1b = (const float*)d_in[11];
  const float* b1b   = (const float*)d_in[12];
  const float* aw1   = (const float*)d_in[13];
  const float* ab1   = (const float*)d_in[14];
  const float* aw2   = (const float*)d_in[15];
  // d_in[16] = ab2: constant shift, drops out of the softmax -> unused.
  const float* cw1   = (const float*)d_in[17];
  const float* cb1   = (const float*)d_in[18];
  const float* cw2   = (const float*)d_in[19];
  const float* cb2   = (const float*)d_in[20];
  (void)in_sizes; (void)n_in; (void)out_size; (void)ws_size;

  float* out_logits = (float*)d_out;
  float* out_attw   = (float*)d_out + BDIM * NCLS;

  char* base = (char*)d_ws;
  size_t off = 0;
  auto take = [&](size_t bytes) -> void* {
    void* p = base + off;
    off = (off + bytes + 255) & ~(size_t)255;
    return p;
  };
  __bf16* xb      = (__bf16*)take((size_t)BT * INPAD * 2);
  __bf16* wih0fb  = (__bf16*)take((size_t)G4 * INPAD * 2);
  __bf16* wih0bb  = (__bf16*)take((size_t)G4 * INPAD * 2);
  __bf16* whh0fb  = (__bf16*)take((size_t)G4 * HDIM * 2);
  __bf16* whh0bb  = (__bf16*)take((size_t)G4 * HDIM * 2);
  __bf16* wih1fb  = (__bf16*)take((size_t)G4 * H2 * 2);
  __bf16* wih1bb  = (__bf16*)take((size_t)G4 * H2 * 2);
  __bf16* whh1fb  = (__bf16*)take((size_t)G4 * HDIM * 2);
  __bf16* whh1bb  = (__bf16*)take((size_t)G4 * HDIM * 2);
  __bf16* aw1T    = (__bf16*)take((size_t)H2 * H2 * 2);
  float*  xpF     = (float*)take((size_t)BT * G4 * 4);   // reused for layer1
  float*  xpB     = (float*)take((size_t)BT * G4 * 4);
  __bf16* h0b     = (__bf16*)take((size_t)BT * H2 * 2);
  float*  lstmF32 = (float*)take((size_t)BT * H2 * 4);
  __bf16* lstmBF  = (__bf16*)take((size_t)BT * H2 * 2);
  __bf16* hbufs   = (__bf16*)take((size_t)2 * 2 * 2 * BDIM * HDIM * 2);  // [layer][buf][dir]
  float*  cstate  = (float*)take((size_t)2 * 2 * BDIM * HDIM * 4);       // [layer][dir]
  float*  scores  = (float*)take((size_t)BT * 4);
  float*  attw    = (float*)take((size_t)BT * 4);
  float*  ctx     = (float*)take((size_t)BDIM * H2 * 4);

  auto g1 = [](size_t n) { return dim3((unsigned)((n + 255) / 256)); };
  const size_t HS = (size_t)BDIM * HDIM;

  // -- state/score init (outputs fully overwritten; ws poisoned by harness) --
  zero_bf16_k<<<g1(8 * HS), 256, 0, stream>>>(hbufs, (int)(8 * HS));
  zero_f32_k<<<g1(4 * HS), 256, 0, stream>>>(cstate, (int)(4 * HS));
  zero_f32_k<<<g1(BT), 256, 0, stream>>>(scores, BT);

  // -- fp32 -> bf16 conversions (K padded to multiple of 32) --
  cvt_pad_k<<<g1((size_t)BT * INPAD), 256, 0, stream>>>(x, xb, BT, INDIM, INPAD);
  cvt_pad_k<<<g1((size_t)G4 * INPAD), 256, 0, stream>>>(wih0f, wih0fb, G4, INDIM, INPAD);
  cvt_pad_k<<<g1((size_t)G4 * INPAD), 256, 0, stream>>>(wih0b, wih0bb, G4, INDIM, INPAD);
  cvt_pad_k<<<g1((size_t)G4 * HDIM), 256, 0, stream>>>(whh0f, whh0fb, G4, HDIM, HDIM);
  cvt_pad_k<<<g1((size_t)G4 * HDIM), 256, 0, stream>>>(whh0b, whh0bb, G4, HDIM, HDIM);
  cvt_pad_k<<<g1((size_t)G4 * H2), 256, 0, stream>>>(wih1f, wih1fb, G4, H2, H2);
  cvt_pad_k<<<g1((size_t)G4 * H2), 256, 0, stream>>>(wih1b, wih1bb, G4, H2, H2);
  cvt_pad_k<<<g1((size_t)G4 * HDIM), 256, 0, stream>>>(whh1f, whh1fb, G4, HDIM, HDIM);
  cvt_pad_k<<<g1((size_t)G4 * HDIM), 256, 0, stream>>>(whh1b, whh1bb, G4, HDIM, HDIM);
  cvt_T_k<<<g1((size_t)H2 * H2), 256, 0, stream>>>(aw1, aw1T, H2, H2);

  // -- layer 0: time-parallel input projections (WMMA) --
  wmma_gemm_k<0><<<dim3(BT / 128, G4 / 128), 256, 0, stream>>>(
      xb, INPAD, wih0fb, INPAD, b0f, xpF, G4, INPAD, nullptr, nullptr);
  wmma_gemm_k<0><<<dim3(BT / 128, G4 / 128), 256, 0, stream>>>(
      xb, INPAD, wih0bb, INPAD, b0b, xpB, G4, INPAD, nullptr, nullptr);

  // -- layer 0 recurrence: fwd+bwd per launch, h double-buffered --
  for (int t = 0; t < TDIM; t++) {
    __bf16* hin  = hbufs + (size_t)(t & 1) * 2 * HS;
    __bf16* hout = hbufs + (size_t)((t + 1) & 1) * 2 * HS;
    lstm_step_k<<<dim3(HDIM / 32, 2), 256, 0, stream>>>(
        xpF, xpB, whh0fb, whh0bb, hin, hout, cstate, h0b, nullptr, t);
  }

  // -- layer 1: input projections from concat(h0f,h0b) bf16 (reuse xp bufs) --
  wmma_gemm_k<0><<<dim3(BT / 128, G4 / 128), 256, 0, stream>>>(
      h0b, H2, wih1fb, H2, b1f, xpF, G4, H2, nullptr, nullptr);
  wmma_gemm_k<0><<<dim3(BT / 128, G4 / 128), 256, 0, stream>>>(
      h0b, H2, wih1bb, H2, b1b, xpB, G4, H2, nullptr, nullptr);

  // -- layer 1 recurrence --
  for (int t = 0; t < TDIM; t++) {
    __bf16* hin  = hbufs + 4 * HS + (size_t)(t & 1) * 2 * HS;
    __bf16* hout = hbufs + 4 * HS + (size_t)((t + 1) & 1) * 2 * HS;
    lstm_step_k<<<dim3(HDIM / 32, 2), 256, 0, stream>>>(
        xpF, xpB, whh1fb, whh1bb, hin, hout, cstate + 2 * HS, lstmBF, lstmF32, t);
  }

  // -- attention: fused WMMA (tanh + aw2 row-reduction in epilogue) --
  wmma_gemm_k<1><<<dim3(BT / 128, H2 / 128), 256, 0, stream>>>(
      lstmBF, H2, aw1T, H2, ab1, nullptr, H2, H2, aw2, scores);
  softmax_k<<<BDIM, TDIM, 0, stream>>>(scores, attw, out_attw);
  context_k<<<BDIM, 256, 0, stream>>>(attw, lstmF32, ctx);
  classifier_k<<<BDIM, 256, 0, stream>>>(ctx, cw1, cb1, cw2, cb2, out_logits);
}
